// FGSW_MSA_7516192768079
// MI455X (gfx1250) — matrix operand order repaired and checked by reading, not compiled, not run on hardware
//
#include <hip/hip_runtime.h>

typedef __attribute__((ext_vector_type(16))) _Float16 v16h;
typedef __attribute__((ext_vector_type(8)))  _Float16 v8h;
typedef __attribute__((ext_vector_type(8)))  float    v8f;
typedef __attribute__((ext_vector_type(4)))  unsigned v4u;

#define HH 256
#define WW 256
#define HW 65536
#define CIN 64
#define HEADS 8
#define DHEAD 64
#define EPS 1e-5f
#define QSCALE 0.125f   // 64^-0.5

union AFrag { v16h h; v4u u4[2]; };

__device__ __forceinline__ void async_lds_b128(unsigned ldsoff, const void* g) {
    asm volatile("global_load_async_to_lds_b128 %0, %1, off"
                 :: "v"(ldsoff), "v"((unsigned long long)g) : "memory");
}
__device__ __forceinline__ void wait_asynccnt0() {
    asm volatile("s_wait_asynccnt 0x0" ::: "memory");
}

// ---------------------------------------------------------------------------
// LayerNorm over channel dim (q path), f32 NCHW -> f16 NHWC
// ---------------------------------------------------------------------------
__global__ void ln_q_kernel(const float* __restrict__ q, const float* __restrict__ g,
                            const float* __restrict__ b, _Float16* __restrict__ out) {
    int p = blockIdx.x * blockDim.x + threadIdx.x;
    if (p >= HW) return;
    float v[CIN];
    float mu = 0.f;
    for (int ic = 0; ic < CIN; ++ic) { v[ic] = q[(long)ic * HW + p]; mu += v[ic]; }
    mu *= (1.f / CIN);
    float var = 0.f;
    for (int ic = 0; ic < CIN; ++ic) { float d = v[ic] - mu; var += d * d; }
    var *= (1.f / CIN);
    float rs = rsqrtf(var + EPS);
    _Float16* o = out + (long)p * CIN;
    for (int ic = 0; ic < CIN; ++ic)
        o[ic] = (_Float16)((v[ic] - mu) * rs * g[ic] + b[ic]);
}

// ---------------------------------------------------------------------------
// Flow warp (nearest, zero pad) + LayerNorm, f32 NCHW -> f16 NHWC per frame
// ---------------------------------------------------------------------------
__global__ void warp_ln_kv_kernel(const float* __restrict__ k_inp,
                                  const float* __restrict__ flow_f,
                                  const float* __restrict__ flow_b,
                                  const float* __restrict__ g,
                                  const float* __restrict__ bb,
                                  _Float16* __restrict__ out) {
    int p = blockIdx.x * blockDim.x + threadIdx.x;
    if (p >= HW) return;
    int f = blockIdx.y;           // 0..2
    int y = p >> 8, x = p & 255;
    const float* src = k_inp + (long)f * CIN * HW;
    long sp = p;
    bool valid = true;
    if (f != 1) {
        const float* fl = (f == 0) ? flow_f : flow_b;
        float gx = (float)x + fl[p];
        float gy = (float)y + fl[HW + p];
        float vx = 2.f * gx / (float)(WW - 1) - 1.f;
        float vy = 2.f * gy / (float)(HH - 1) - 1.f;
        float ixf = rintf((vx + 1.f) * (0.5f * WW) - 0.5f);
        float iyf = rintf((vy + 1.f) * (0.5f * HH) - 0.5f);
        valid = (ixf >= 0.f) && (ixf < (float)WW) && (iyf >= 0.f) && (iyf < (float)HH);
        int ix = min(max((int)ixf, 0), WW - 1);
        int iy = min(max((int)iyf, 0), HH - 1);
        sp = (long)iy * WW + ix;
    }
    float v[CIN];
    float mu = 0.f;
    for (int ic = 0; ic < CIN; ++ic) {
        float t = valid ? src[(long)ic * HW + sp] : 0.f;
        v[ic] = t; mu += t;
    }
    mu *= (1.f / CIN);
    float var = 0.f;
    for (int ic = 0; ic < CIN; ++ic) { float d = v[ic] - mu; var += d * d; }
    var *= (1.f / CIN);
    float rs = rsqrtf(var + EPS);
    _Float16* o = out + ((long)f * HW + p) * CIN;
    for (int ic = 0; ic < CIN; ++ic)
        o[ic] = (_Float16)((v[ic] - mu) * rs * g[ic] + bb[ic]);
}

// ---------------------------------------------------------------------------
// Weight pack: f32 [oc][ic][3][3] -> f16 [oc][k], k = (dy*3+dx)*C_in + ic
// ---------------------------------------------------------------------------
__global__ void pack_w_kernel(const float* __restrict__ src, _Float16* __restrict__ dst,
                              int C_in, int n) {
    int i = blockIdx.x * blockDim.x + threadIdx.x;
    if (i >= n) return;
    int K = C_in * 9;
    int oc = i / K, rem = i - oc * K;
    int ic = rem / 9, j9 = rem - ic * 9;
    dst[(long)oc * K + j9 * C_in + ic] = (_Float16)src[i];
}

// ---------------------------------------------------------------------------
// 3x3 conv as implicit GEMM on v_wmma_f32_16x16x32_f16, NHWC activations.
// Block = 256 threads (8 waves) computes a 32-pixel row segment:
//   patch [3][34][CI] staged once with 16B global_load_async_to_lds_b128.
// Waves split 8 = (8/NWN along M) x (NWN along N); each wave owns MT M-tiles
// and 2/NWN N-tiles, so every A fragment feeds 2/NWN WMMAs.
// MODE 1: f32 NCHW out.  MODE 2: windowed Q f16.  MODE 3: windowed K + V^T f16.
// ---------------------------------------------------------------------------
template <int MODE, int MT, int CI, int NWN>
__global__ void conv3x3_wmma_kernel(const _Float16* __restrict__ in,
                                    const _Float16* __restrict__ wt,
                                    void* __restrict__ out0,
                                    void* __restrict__ out1,
                                    int C_out, float scale, long in_fstride) {
    extern __shared__ _Float16 patch[];   // 3 * 34 * CI halfs
    const int tile = blockIdx.x;          // 256 rows x 8 tiles of 32 px
    const int y  = tile >> 3;
    const int x0 = (tile & 7) << 5;
    const int frame = blockIdx.z;
    in += (long)frame * in_fstride;
    const int tid = threadIdx.x;

    // ---- async-stage NHWC patch (rows y-1..y+1, cols x0-1..x0+32) ----
    constexpr int VPC = CI / 8;                  // b128 transfers per column
    constexpr int NSEG = 3 * 34 * VPC;
    for (int i = tid; i < NSEG; i += blockDim.x) {
        int ic8 = i % VPC;
        int cc  = (i / VPC) % 34;
        int r   = i / (34 * VPC);
        int gy = y - 1 + r, gx = x0 - 1 + cc;
        int el = ((r * 34 + cc) * CI) + ic8 * 8;
        if (gy >= 0 && gy < HH && gx >= 0 && gx < WW) {
            unsigned lo = (unsigned)(unsigned long long)&patch[el];
            async_lds_b128(lo, in + ((long)gy * WW + gx) * CI + ic8 * 8);
        } else {
            *reinterpret_cast<v4u*>(&patch[el]) = (v4u){0u, 0u, 0u, 0u};
        }
    }
    wait_asynccnt0();
    __syncthreads();

    const int lane = tid & 31;
    const int wave = tid >> 5;
    const int n  = lane & 15;
    const int hi = lane >> 4;
    constexpr int K   = CI * 9;
    constexpr int KC  = K >> 5;
    constexpr int MW  = 8 / NWN;     // wave-groups along M
    constexpr int NTW = 2 / NWN;     // N-tiles per wave
    const int mgrp = wave & (MW - 1);
    const int ngrp = wave / MW;      // 0 .. NWN-1
    const int total_mt = C_out >> 4;
    const int gw = blockIdx.y * MW + mgrp;

    v8f acc[MT][NTW];
    bool act[MT];
#pragma unroll
    for (int mi = 0; mi < MT; ++mi) {
        act[mi] = (gw * MT + mi) < total_mt;
#pragma unroll
        for (int ni = 0; ni < NTW; ++ni)
            acc[mi][ni] = (v8f){0.f, 0.f, 0.f, 0.f, 0.f, 0.f, 0.f, 0.f};
    }

    for (int kc = 0; kc < KC; ++kc) {
        // B fragments: 16 contiguous channels at one (dy,dx,col) -> one v16h
        const int k16 = (kc << 5) + (hi ? 16 : 0);
        const int j9  = k16 / CI;
        const int icb = k16 - j9 * CI;
        const int dy = j9 / 3, dx = j9 - dy * 3;
        v16h bfrag[NTW];
#pragma unroll
        for (int ni = 0; ni < NTW; ++ni) {
            int col = (ngrp * NTW + ni) * 16 + n + dx;
            bfrag[ni] = *(const v16h*)&patch[((dy * 34 + col) * CI) + icb];
        }
        // A fragments: two b128 loads per M-tile, reused across N-tiles
#pragma unroll
        for (int mi = 0; mi < MT; ++mi) {
            if (!act[mi]) continue;
            const v4u* wrowq = (const v4u*)(wt + (long)(((gw * MT + mi) << 4) + n) * K);
            if (kc + 1 < KC) __builtin_prefetch(wrowq + 4 * (kc + 1), 0, 1);
            AFrag af;
            af.u4[0] = wrowq[4 * kc + 2 * hi];
            af.u4[1] = wrowq[4 * kc + 2 * hi + 1];
#pragma unroll
            for (int ni = 0; ni < NTW; ++ni)
                acc[mi][ni] = __builtin_amdgcn_wmma_f32_16x16x32_f16(
                    false, af.h, false, bfrag[ni], (short)0, acc[mi][ni], false, false);
        }
    }

#pragma unroll
    for (int ni = 0; ni < NTW; ++ni) {
        const int px = x0 + (ngrp * NTW + ni) * 16 + n;
        const int win = (y >> 2) * 64 + (px >> 2);
        const int token = (y & 3) * 4 + (px & 3);
#pragma unroll
        for (int mi = 0; mi < MT; ++mi) {
            if (!act[mi]) continue;
            int ocb = ((gw * MT + mi) << 4) + (hi ? 8 : 0);
            if (MODE == 1) {
                float* o = (float*)out0;
#pragma unroll
                for (int r = 0; r < 8; ++r)
                    o[(long)(ocb + r) * HW + (long)y * WW + px] = acc[mi][ni][r] * scale;
            } else if (MODE == 2) {
                v8h pk;
#pragma unroll
                for (int r = 0; r < 8; ++r) pk[r] = (_Float16)(acc[mi][ni][r] * scale);
                int head = ocb >> 6, d0 = ocb & 63;
                _Float16* o = (_Float16*)out0 + (((long)win * 8 + head) * 16 + token) * 64 + d0;
                *(v8h*)o = pk;
            } else {  // MODE 3: K windowed + V transposed
                int ocm = ocb & 511;
                int head = ocm >> 6, d0 = ocm & 63;
                int key = frame * 16 + token;
                if (ocb < 512) {
                    v8h pk;
#pragma unroll
                    for (int r = 0; r < 8; ++r) pk[r] = (_Float16)(acc[mi][ni][r] * scale);
                    _Float16* o = (_Float16*)out0 + (((long)win * 8 + head) * 48 + key) * 64 + d0;
                    *(v8h*)o = pk;
                } else {
                    _Float16* o = (_Float16*)out1 + (((long)win * 8 + head) * 64 + d0) * 48 + key;
#pragma unroll
                    for (int r = 0; r < 8; ++r) o[(long)r * 48] = (_Float16)(acc[mi][ni][r] * scale);
                }
            }
        }
    }
}

// ---------------------------------------------------------------------------
// Windowed attention. Block = 256 = 8 waves, wave = head.
// K rows / V^T rows contiguous -> whole B fragments load as v16h.
// ---------------------------------------------------------------------------
__global__ void attn_wmma_kernel(const _Float16* __restrict__ qwin,
                                 const _Float16* __restrict__ kwin,
                                 const _Float16* __restrict__ vtwin,
                                 const float* __restrict__ static_a,
                                 _Float16* __restrict__ ow) {
    __shared__ float    simlds[HEADS * 16 * 48];
    __shared__ _Float16 attnlds[HEADS * 16 * 64];
    const int win = blockIdx.x;
    const int wy = win >> 6, wx = win & 63;
    const int y0 = wy << 2, x0 = wx << 2;
    const int tid  = threadIdx.x;
    const int head = tid >> 5;
    const int lane = tid & 31;
    const int n  = lane & 15;
    const int hi = lane >> 4;

    // ---- q A fragments: two b128 loads per chunk ----
    AFrag aq[2];
    {
        const v4u* qrow = (const v4u*)(qwin + (((long)win * 8 + head) * 16 + n) * 64);
#pragma unroll
        for (int c = 0; c < 2; ++c) {
            aq[c].u4[0] = qrow[c * 4 + 2 * hi];
            aq[c].u4[1] = qrow[c * 4 + 2 * hi + 1];
        }
    }

    // ---- sim = q . k^T over 3 key tiles ----
    const _Float16* kbase = kwin + ((long)win * 8 + head) * 48 * 64;
    for (int t = 0; t < 3; ++t) {
        v8f acc = (v8f){0.f, 0.f, 0.f, 0.f, 0.f, 0.f, 0.f, 0.f};
        int key = t * 16 + n;
#pragma unroll
        for (int c = 0; c < 2; ++c) {
            v16h bf = *(const v16h*)(kbase + (long)key * 64 + c * 32 + hi * 16);
            acc = __builtin_amdgcn_wmma_f32_16x16x32_f16(
                false, aq[c].h, false, bf, (short)0, acc, false, false);
        }
        int mrow = hi ? 8 : 0;
#pragma unroll
        for (int r = 0; r < 8; ++r) {
            int m = mrow + r;
            simlds[(head * 16 + m) * 48 + key] = acc[r] + static_a[(head * 16 + m) * 48 + key];
        }
    }
    __syncthreads();

    // ---- softmax over 48 keys, attn -> f16 LDS padded to 64 ----
    if (lane < 16) {
        int m = lane;
        float* row = &simlds[(head * 16 + m) * 48];
        float mx = row[0];
        for (int j = 1; j < 48; ++j) mx = fmaxf(mx, row[j]);
        float s = 0.f;
        for (int j = 0; j < 48; ++j) { float e = __expf(row[j] - mx); row[j] = e; s += e; }
        float inv = 1.f / s;
        _Float16* arow = &attnlds[(head * 16 + m) * 64];
        for (int j = 0; j < 48; ++j) arow[j] = (_Float16)(row[j] * inv);
        for (int j = 48; j < 64; ++j) arow[j] = (_Float16)0.f;
    }
    __syncthreads();

    // ---- attn A fragments from LDS (b128 reads) ----
    AFrag aa[2];
    {
        const v4u* arowq = (const v4u*)&attnlds[(head * 16 + n) * 64];
#pragma unroll
        for (int c = 0; c < 2; ++c) {
            aa[c].u4[0] = arowq[c * 4 + 2 * hi];
            aa[c].u4[1] = arowq[c * 4 + 2 * hi + 1];
        }
    }

    // ---- out = attn . v via V^T rows (contiguous v16h loads) ----
    const _Float16* vbase = vtwin + ((long)win * 8 + head) * 64 * 48;
    for (int dt = 0; dt < 4; ++dt) {
        int d = dt * 16 + n;
        v8f acc = (v8f){0.f, 0.f, 0.f, 0.f, 0.f, 0.f, 0.f, 0.f};
#pragma unroll
        for (int c = 0; c < 2; ++c) {
            int cb = c * 32 + hi * 16;     // key base: 0,16,32,48
            v16h bf;
            if (cb == 48) {
#pragma unroll
                for (int e = 0; e < 16; ++e) bf[e] = (_Float16)0.f;   // zero-pad keys 48..63
            } else {
                bf = *(const v16h*)(vbase + (long)d * 48 + cb);
            }
            acc = __builtin_amdgcn_wmma_f32_16x16x32_f16(
                false, aa[c].h, false, bf, (short)0, acc, false, false);
        }
        int ch = head * DHEAD + d;    // contiguous across the half-wave
#pragma unroll
        for (int r = 0; r < 8; ++r) {
            int m = (hi ? 8 : 0) + r;
            int py = y0 + (m >> 2), px = x0 + (m & 3);
            ow[((long)py * WW + px) * 512 + ch] = (_Float16)acc[r];   // NHWC
        }
    }
}

// ---------------------------------------------------------------------------
extern "C" void kernel_launch(void* const* d_in, const int* in_sizes, int n_in,
                              void* d_out, int out_size, void* d_ws, size_t ws_size,
                              hipStream_t stream) {
    (void)in_sizes; (void)n_in; (void)out_size; (void)ws_size;
    const float* q_inp    = (const float*)d_in[0];
    const float* k_inp    = (const float*)d_in[1];
    const float* flow_f   = (const float*)d_in[2];
    const float* flow_b   = (const float*)d_in[3];
    const float* static_a = (const float*)d_in[4];
    const float* ln_q_g   = (const float*)d_in[5];
    const float* ln_q_b   = (const float*)d_in[6];
    const float* ln_kv_g  = (const float*)d_in[7];
    const float* ln_kv_b  = (const float*)d_in[8];
    const float* W_q      = (const float*)d_in[9];
    const float* W_kv     = (const float*)d_in[10];
    const float* W_out    = (const float*)d_in[11];

    char* ws = (char*)d_ws;
    _Float16* qn    = (_Float16*)ws; ws += (size_t)CIN * HW * 2;       // LN'd q, NHWC
    _Float16* kn    = (_Float16*)ws; ws += (size_t)3 * CIN * HW * 2;   // warped+LN'd kv, NHWC
    _Float16* wqh   = (_Float16*)ws; ws += (size_t)512 * 576 * 2;
    _Float16* wkvh  = (_Float16*)ws; ws += (size_t)1024 * 576 * 2;
    _Float16* wouth = (_Float16*)ws; ws += (size_t)64 * 4608 * 2;
    _Float16* qwin  = (_Float16*)ws; ws += (size_t)512 * HW * 2;       // [win][head][16][64]
    _Float16* kwin  = (_Float16*)ws; ws += (size_t)3 * 512 * HW * 2;   // [win][head][48][64]
    _Float16* vtwin = (_Float16*)ws; ws += (size_t)3 * 512 * HW * 2;   // [win][head][64][48]
    _Float16* owh   = (_Float16*)ws; ws += (size_t)512 * HW * 2;       // attn out, NHWC

    ln_q_kernel<<<HW / 256, 256, 0, stream>>>(q_inp, ln_q_g, ln_q_b, qn);
    warp_ln_kv_kernel<<<dim3(HW / 256, 3), 256, 0, stream>>>(
        k_inp, flow_f, flow_b, ln_kv_g, ln_kv_b, kn);

    pack_w_kernel<<<(512 * 576 + 255) / 256, 256, 0, stream>>>(W_q, wqh, 64, 512 * 576);
    pack_w_kernel<<<(1024 * 576 + 255) / 256, 256, 0, stream>>>(W_kv, wkvh, 64, 1024 * 576);
    pack_w_kernel<<<(64 * 4608 + 255) / 256, 256, 0, stream>>>(W_out, wouth, 512, 64 * 4608);

    // Q: 32 M-tiles = 8 M-wavegroups x MT4; 2 N-tiles per wave; 32-px blocks
    conv3x3_wmma_kernel<2, 4, 64, 1><<<dim3(2048, 1, 1), 256, 3 * 34 * 64 * 2, stream>>>(
        qn, wqh, qwin, nullptr, 512, QSCALE, 0);
    // KV: 64 M-tiles = 8 M-wavegroups x MT8; one staging per patch; frames on z
    conv3x3_wmma_kernel<3, 8, 64, 1><<<dim3(2048, 1, 3), 256, 3 * 34 * 64 * 2, stream>>>(
        kn, wkvh, kwin, vtwin, 1024, 1.0f, (long)CIN * HW);

    attn_wmma_kernel<<<4096, 256, 0, stream>>>(qwin, kwin, vtwin, static_a, owh);

    // Out: 4 M-tiles; waves split 4xM x 2xN so all 8 waves are active; f32 NCHW
    conv3x3_wmma_kernel<1, 1, 512, 2><<<dim3(2048, 1, 1), 256, 3 * 34 * 512 * 2, stream>>>(
        owh, wouth, d_out, nullptr, 64, 1.0f, 0);
}
